// Gemma3nTextAltUp_87789131530454
// MI455X (gfx1250) — matrix operand order repaired and checked by reading, not compiled, not run on hardware
//
#include <hip/hip_runtime.h>

typedef __attribute__((ext_vector_type(16))) _Float16 v16h;
typedef __attribute__((ext_vector_type(8)))  float    v8f;
typedef __attribute__((ext_vector_type(4)))  float    v4f;

namespace {
constexpr int   kH   = 2048;   // hidden size
constexpr int   kN   = 4;      // modalities
constexpr float kEps = 1e-6f;
}

// ---------------------------------------------------------------------------
// Kernel 1: per-token router. One wave = 16 tokens. Computes
//   m  = tanh(rmsnorm(hs0 , w)/H @ router_w^T)   -> coefs[16] via pred_w
//   m2 = tanh(rmsnorm(act , w)/H @ router_w^T)   -> ccoef[4]  via corr_w (+1)
// The two 16x2048 @ 2048x16 dot-product batches run on v_wmma_f32_16x16x32_f16.
// ---------------------------------------------------------------------------
__global__ __launch_bounds__(256) void altup_router_kernel(
    const float* __restrict__ hs,        // [N,T,H]  (plane 0 used)
    const float* __restrict__ act,       // [T,H]
    const float* __restrict__ wnorm,     // [H]
    const float* __restrict__ router_w,  // [N,H]
    const float* __restrict__ pred_w,    // [16,4]
    const float* __restrict__ corr_w,    // [4,4]
    float* __restrict__ coef_ws,         // [T,20]: 16 pred coefs + 4 corr coefs
    int T)
{
  __shared__ float m_lds [8][16][4];
  __shared__ float m2_lds[8][16][4];

  const int lane = threadIdx.x & 31;
  const int wave = threadIdx.x >> 5;
  int t0 = (blockIdx.x * 8 + wave) * 16;
  if (t0 + 16 > T) t0 = T - 16;          // clamp (T assumed >= 16)

  const int row  = lane & 15;            // A: token row   / B,C: column n
  const int half = lane >> 4;

  const float* hs0row = hs  + (size_t)(t0 + row) * kH;
  const float* actrow = act + (size_t)(t0 + row) * kH;

  v8f acc1 = {};   // hs0 @ (router*wnorm)^T
  v8f acc2 = {};   // act @ (router*wnorm)^T
  float ss1 = 0.f, ss2 = 0.f;            // partial sum-of-squares (f32)

  for (int kb = 0; kb < kH; kb += 32) {
    // ---- A tiles (16-bit A 16x32 layout, ISA 7.12.2) ----
    const int o1 = kb + half * 8;        // K  o1..o1+7  -> a[0..7]
    const int o2 = kb + 16 + half * 8;   // K  o2..o2+7  -> a[8..15]
    v16h a1, a2;
#pragma unroll
    for (int j = 0; j < 8; ++j) {
      float x = hs0row[o1 + j];
      float y = hs0row[o2 + j];
      ss1 += x * x + y * y;
      a1[j] = (_Float16)x;  a1[8 + j] = (_Float16)y;
      float u = actrow[o1 + j];
      float v = actrow[o2 + j];
      ss2 += u * u + v * v;
      a2[j] = (_Float16)u;  a2[8 + j] = (_Float16)v;
    }
    // ---- B tile: (router_w * wnorm), cols n>=4 are zero ----
    const int kbB = kb + half * 8;       // lanes<16: K 0..7,16..23, lanes>=16: K 8..15,24..31
    v16h b;
    if (row < kN) {
      const float* rw = router_w + (size_t)row * kH + kbB;
      const float* wn = wnorm + kbB;
#pragma unroll
      for (int j = 0; j < 16; ++j) { const int kk = j + (j >> 3) * 8; b[j] = (_Float16)(rw[kk] * wn[kk]); }
    } else {
#pragma unroll
      for (int j = 0; j < 16; ++j) b[j] = (_Float16)0.f;
    }
    acc1 = __builtin_amdgcn_wmma_f32_16x16x32_f16(false, a1, false, b,
                                                  (short)0, acc1, false, false);
    acc2 = __builtin_amdgcn_wmma_f32_16x16x32_f16(false, a2, false, b,
                                                  (short)0, acc2, false, false);
  }

  // complete per-token sum-of-squares: partner lane holds the other K half
  ss1 += __shfl_xor(ss1, 16, 32);
  ss2 += __shfl_xor(ss2, 16, 32);
  // lane L now holds ss for token (L & 15)

  const float inv_h = 1.0f / (float)kH;
  const int n = row;                     // this lane's C column
#pragma unroll
  for (int v = 0; v < 8; ++v) {
    const int tv = v + 8 * half;         // C row (token index within group)
    float s1 = __shfl(ss1, tv, 32);
    float s2 = __shfl(ss2, tv, 32);
    float r1 = rsqrtf(s1 * inv_h + kEps);
    float r2 = rsqrtf(s2 * inv_h + kEps);
    if (n < kN) {
      m_lds [wave][tv][n] = tanhf(acc1[v] * r1 * inv_h);
      m2_lds[wave][tv][n] = tanhf(acc2[v] * r2 * inv_h);
    }
  }
  __syncthreads();

  if (half == 0) {                       // lanes 0..15: one token each
    const int t = t0 + row;
    float m[4], m2[4];
#pragma unroll
    for (int i = 0; i < 4; ++i) { m[i] = m_lds[wave][row][i]; m2[i] = m2_lds[wave][row][i]; }
    float* outp = coef_ws + (size_t)t * 20;
#pragma unroll
    for (int k = 0; k < 16; ++k) {       // coefs[j,i] = flat[j*4+i]
      float c = 0.f;
#pragma unroll
      for (int i = 0; i < 4; ++i) c += m[i] * pred_w[k * 4 + i];
      outp[k] = c;
    }
#pragma unroll
    for (int j = 0; j < 4; ++j) {        // ccoef[j] = 1 + m2 . corr_w[j,:]
      float c = 1.0f;
#pragma unroll
      for (int i = 0; i < 4; ++i) c += m2[i] * corr_w[j * 4 + i];
      outp[16 + j] = c;
    }
  }
}

// ---------------------------------------------------------------------------
// Kernel 2: streaming apply. One block per token, 512 threads x float4 = 2048.
//   pred[j] = hs_j + sum_i hs_i * coef[j*4+i]
//   out[j]  = ((act - pred[0]) * ccoef[j] + pred[j]) * scale
// ---------------------------------------------------------------------------
__global__ __launch_bounds__(512) void altup_apply_kernel(
    const float* __restrict__ hs,        // [N,T,H]
    const float* __restrict__ act,       // [T,H]
    const float* __restrict__ oscale,    // [H]
    const float* __restrict__ coef_ws,   // [T,20]
    float* __restrict__ out,             // [N,T,H]
    int T)
{
  const int t = blockIdx.x;
  const size_t plane = (size_t)T * kH;
  const size_t base  = (size_t)t * kH + (size_t)threadIdx.x * 4;

  const float* cp = coef_ws + (size_t)t * 20;   // uniform per block
  float c[20];
#pragma unroll
  for (int k = 0; k < 20; ++k) c[k] = cp[k];

  float xv[4][4];
#pragma unroll
  for (int i = 0; i < 4; ++i) {
    v4f v4 = *reinterpret_cast<const v4f*>(hs + (size_t)i * plane + base);
#pragma unroll
    for (int e = 0; e < 4; ++e) xv[i][e] = v4[e];
  }
  float av[4], sv[4];
  {
    v4f a4 = *reinterpret_cast<const v4f*>(act + base);
    v4f s4 = *reinterpret_cast<const v4f*>(oscale + (size_t)threadIdx.x * 4);
#pragma unroll
    for (int e = 0; e < 4; ++e) { av[e] = a4[e]; sv[e] = s4[e]; }
  }

  float ov[4][4];
#pragma unroll
  for (int e = 0; e < 4; ++e) {          // vector component
    float pred[4];
#pragma unroll
    for (int j = 0; j < 4; ++j) {
      float p = xv[j][e];
#pragma unroll
      for (int i = 0; i < 4; ++i) p += xv[i][e] * c[j * 4 + i];
      pred[j] = p;
    }
    const float innov = av[e] - pred[0];
#pragma unroll
    for (int j = 0; j < 4; ++j)
      ov[j][e] = (innov * c[16 + j] + pred[j]) * sv[e];
  }
#pragma unroll
  for (int j = 0; j < 4; ++j) {
    v4f o4 = { ov[j][0], ov[j][1], ov[j][2], ov[j][3] };
    __builtin_nontemporal_store(o4, reinterpret_cast<v4f*>(out + (size_t)j * plane + base));
  }
}

// ---------------------------------------------------------------------------
extern "C" void kernel_launch(void* const* d_in, const int* in_sizes, int n_in,
                              void* d_out, int out_size, void* d_ws, size_t ws_size,
                              hipStream_t stream) {
  const float* hs     = (const float*)d_in[0];  // [N,B,S,H]
  const float* act    = (const float*)d_in[1];  // [B,S,H]
  const float* wnorm  = (const float*)d_in[2];  // [H]
  const float* rw     = (const float*)d_in[3];  // [N,H]
  const float* pw     = (const float*)d_in[4];  // [16,4]
  const float* cw     = (const float*)d_in[5];  // [4,4]
  const float* osc    = (const float*)d_in[6];  // [H]
  float*       out    = (float*)d_out;          // [N,B,S,H]
  float*       coefws = (float*)d_ws;           // [T,20]

  const int T = in_sizes[1] / kH;               // B*S tokens
  const int waves   = (T + 15) / 16;
  const int blocks1 = (waves + 7) / 8;          // 8 waves (256 thr) per block

  altup_router_kernel<<<blocks1, 256, 0, stream>>>(hs, act, wnorm, rw, pw, cw,
                                                   coefws, T);
  altup_apply_kernel<<<T, 512, 0, stream>>>(hs, act, osc, coefws, out, T);
}
